// ChannelAttention_68547678044203
// MI455X (gfx1250) — matrix operand order repaired and checked by reading, not compiled, not run on hardware
//
#include <hip/hip_runtime.h>
#include <hip/hip_bf16.h>

typedef __bf16 bf16;
typedef __attribute__((ext_vector_type(16))) __bf16 v16bf;
typedef __attribute__((ext_vector_type(8)))  __bf16 v8bf;
typedef __attribute__((ext_vector_type(4)))  __bf16 v4bf;
typedef __attribute__((ext_vector_type(8)))  float  v8f;
typedef __attribute__((ext_vector_type(4)))  float  v4f;
typedef __attribute__((ext_vector_type(4)))  int    v4i;

#define DIM_B 8
#define DIM_L 4096
#define DIM_C 512
#define DIM_3C 1536
#define GROUPS 8
#define HD 64
#define M_TOTAL (DIM_B * DIM_L)   // 32768
#define TPAD 136                  // attn transposed-tile stride (272B = 17*16B)
#define GPAD 40                   // gemm LDS tile row stride  (80B = 5*16B)

#define AS1 __attribute__((address_space(1)))
#define AS3 __attribute__((address_space(3)))

#if defined(__has_builtin) && __has_builtin(__builtin_amdgcn_global_load_async_to_lds_b128)
#define USE_ASYNC_BUILTIN 1
#else
#define USE_ASYNC_BUILTIN 0
#endif

#if defined(__has_builtin) && __has_builtin(__builtin_amdgcn_s_wait_asynccnt)
#define USE_WAITASYNC_BUILTIN 1
#else
#define USE_WAITASYNC_BUILTIN 0
#endif

// 16B global -> LDS async copy (ASYNCcnt-tracked, per-lane addresses)
__device__ __forceinline__ void async_copy_b128(const bf16* __restrict__ g, bf16* l) {
#if USE_ASYNC_BUILTIN
  __builtin_amdgcn_global_load_async_to_lds_b128(
      (AS1 v4i*)g, (AS3 v4i*)l, 0, 0);
#else
  asm volatile("global_load_async_to_lds_b128 %0, %1, off"
               :: "v"((unsigned)(unsigned long long)l), "v"(g) : "memory");
#endif
}

template <int N>
__device__ __forceinline__ void wait_asynccnt() {
#if USE_WAITASYNC_BUILTIN
  __builtin_amdgcn_s_wait_asynccnt((short)N);
#else
  asm volatile("s_wait_asynccnt %0" :: "i"(N) : "memory");
#endif
}

// ---------------------------------------------------------------------------
// WMMA wrapper: D(f32 16x16) = A(bf16 16x32) * B(bf16 32x16) + C
// ---------------------------------------------------------------------------
__device__ __forceinline__ v8f wmma_bf16(v16bf a, v16bf b, v8f c) {
  return __builtin_amdgcn_wmma_f32_16x16x32_bf16(
      /*neg_a=*/false, a, /*neg_b=*/false, b,
      /*c_mod=*/(short)0, c, /*reuse_a=*/false, /*reuse_b=*/false);
}

__device__ __forceinline__ v16bf pack_frag(v4f x, v4f y) {
  v8f t;
#pragma unroll
  for (int i = 0; i < 4; ++i) { t[i] = x[i]; t[i + 4] = y[i]; }
  return __builtin_bit_cast(v16bf, t);
}

// A fragment (16x32 bf16) from row-major A[m*lda + k].
// lanes 0-15 = M rows, hi=lane>>4; elems 0..7 -> K=hi*8+0..7,
// elems 8..15 -> K=16+hi*8+0..7  ==> two 16B loads at +hi*8 and +16+hi*8.
__device__ __forceinline__ v16bf load_fragA(const bf16* __restrict__ base, int lda) {
  const int lane = threadIdx.x & 31;
  const bf16* p = base + (size_t)(lane & 15) * lda + (lane >> 4) * 8;
  return pack_frag(*(const v4f*)p, *(const v4f*)(p + 16));
}

// B fragment (32x16 bf16) from K-contiguous storage Bt[n*ldb + k]
// (logical B[k,n] = Bt[n,k]).  Lanes 0-15 = N cols, elems 0..7 -> K=hi*8+0..7, elems 8..15 -> K=16+hi*8+0..7.
__device__ __forceinline__ v16bf load_fragB(const bf16* __restrict__ Bt, int ldb) {
  const int lane = threadIdx.x & 31;
  const bf16* p = Bt + (size_t)(lane & 15) * ldb + (lane >> 4) * 8;
  return pack_frag(*(const v4f*)p, *(const v4f*)(p + 16));
}

// ---------------------------------------------------------------------------
// Precision-conversion kernels
// ---------------------------------------------------------------------------
__global__ __launch_bounds__(256) void cvt_f32_bf16_v4(const float* __restrict__ in,
                                                       bf16* __restrict__ out, int n4) {
  int i = blockIdx.x * blockDim.x + threadIdx.x;
  int stride = gridDim.x * blockDim.x;
  for (; i < n4; i += stride) {
    v4f x = *(const v4f*)&in[i * 4];
    v4bf o;
#pragma unroll
    for (int j = 0; j < 4; ++j) o[j] = (bf16)x[j];
    *(v4bf*)&out[i * 4] = o;
  }
}

// out[n*K + k] = (bf16) in[k*N + n]  (weights -> K-contiguous bf16)
__global__ __launch_bounds__(256) void transpose_cvt(const float* __restrict__ in,
                                                     bf16* __restrict__ out, int K, int N) {
  int i = blockIdx.x * blockDim.x + threadIdx.x;
  int total = K * N;
  int stride = gridDim.x * blockDim.x;
  for (; i < total; i += stride) {
    int k = i / N, n = i - k * N;
    out[(size_t)n * K + k] = (bf16)in[i];
  }
}

__global__ void copy_size_kernel(const int* __restrict__ in, int* __restrict__ out) {
  out[0] = in[0];
}

// ---------------------------------------------------------------------------
// bf16 GEMM: D[M,N] = A[M,K] * Bt[N,K]^T + bias[N]
// 256 threads = 8 waves (2 x 4); wave tile 32x64; block tile 64x256; K-step 32.
// A/B K-tiles staged through double-buffered LDS with async global->LDS copies
// (ASYNCcnt): issue next tile, s_wait_asynccnt for oldest, barrier, compute.
// ---------------------------------------------------------------------------
template <bool OUT_BF16>
__global__ __launch_bounds__(256) void gemm_bf16_kernel(
    const bf16* __restrict__ A, const bf16* __restrict__ Bt,
    const float* __restrict__ bias, bf16* __restrict__ outH,
    float* __restrict__ outF, int M, int N, int K) {
  __shared__ bf16 Abuf[2][64 * GPAD];    // [row][k] padded, 5,120B each
  __shared__ bf16 Bbuf[2][256 * GPAD];   // [col][k] padded, 20,480B each

  const int tid = threadIdx.x;
  const int wave = tid >> 5;
  const int wm = wave & 1;          // 0..1
  const int wn = wave >> 1;         // 0..3
  const int brow = blockIdx.x * 64;
  const int bcol = blockIdx.y * 256;
  const int row0 = brow + wm * 32;
  const int col0 = bcol + wn * 64;

  // per-thread async-copy assignments (16B chunks)
  const int ar = tid >> 2;          // A: row 0..63
  const int ac = (tid & 3) * 8;     // A: k  0,8,16,24

  v8f acc[2][4];
#pragma unroll
  for (int mi = 0; mi < 2; ++mi)
#pragma unroll
    for (int ni = 0; ni < 4; ++ni) acc[mi][ni] = (v8f)(0.0f);

  const int NT = K >> 5;   // K / 32 tiles

  // issue tile 0
  async_copy_b128(A + (size_t)(brow + ar) * K + ac, &Abuf[0][ar * GPAD + ac]);
#pragma unroll
  for (int i = 0; i < 4; ++i) {
    int id = tid + i * 256;
    int nr = id >> 2, nc = (id & 3) * 8;
    async_copy_b128(Bt + (size_t)(bcol + nr) * K + nc, &Bbuf[0][nr * GPAD + nc]);
  }

  for (int kt = 0; kt < NT; ++kt) {
    const int cur = kt & 1;
    if (kt + 1 < NT) {               // prefetch next K-tile into other buffer
      const int nxt = cur ^ 1;
      const int kof = (kt + 1) << 5;
      async_copy_b128(A + (size_t)(brow + ar) * K + kof + ac,
                      &Abuf[nxt][ar * GPAD + ac]);
#pragma unroll
      for (int i = 0; i < 4; ++i) {
        int id = tid + i * 256;
        int nr = id >> 2, nc = (id & 3) * 8;
        async_copy_b128(Bt + (size_t)(bcol + nr) * K + kof + nc,
                        &Bbuf[nxt][nr * GPAD + nc]);
      }
      wait_asynccnt<5>();            // oldest 5 (current buffer) complete
    } else {
      wait_asynccnt<0>();
    }
    __syncthreads();

    v16bf a0 = load_fragA(&Abuf[cur][(wm * 32) * GPAD], GPAD);
    v16bf a1 = load_fragA(&Abuf[cur][(wm * 32 + 16) * GPAD], GPAD);
#pragma unroll
    for (int ni = 0; ni < 4; ++ni) {
      v16bf b = load_fragB(&Bbuf[cur][(wn * 64 + ni * 16) * GPAD], GPAD);
      acc[0][ni] = wmma_bf16(a0, b, acc[0][ni]);
      acc[1][ni] = wmma_bf16(a1, b, acc[1][ni]);
    }
    __syncthreads();
  }

  const int lane = tid & 31;
  const int n = lane & 15;
  const int hi = lane >> 4;
#pragma unroll
  for (int mi = 0; mi < 2; ++mi)
#pragma unroll
    for (int ni = 0; ni < 4; ++ni) {
      int col = col0 + ni * 16 + n;
      float bv = bias[col];
#pragma unroll
      for (int r = 0; r < 8; ++r) {
        int row = row0 + mi * 16 + hi * 8 + r;
        float val = acc[mi][ni][r] + bv;
        if (OUT_BF16) outH[(size_t)row * N + col] = (bf16)val;
        else          outF[(size_t)row * N + col] = val;
      }
    }
}

// ---------------------------------------------------------------------------
// Channel (cross-covariance) attention core.  One block per (b, g).
// S[d,e] = scale * sum_l q[l,d] k[l,e]  ->  softmax rows  ->
// out[l,d] = sum_e attn[d,e] v[l,e]   (stored bf16 at [b, l, g*64+d])
// q/k tiles staged into LDS *transposed* so every WMMA fragment is two
// aligned ds_load_b128 (row stride 272B: 16B multiple, conflict-free reads).
// ---------------------------------------------------------------------------
__global__ __launch_bounds__(256) void attn_kernel(const bf16* __restrict__ QKV,
                                                   bf16* __restrict__ OutAtt) {
  const int b = blockIdx.x >> 3;
  const int g = blockIdx.x & 7;
  const bf16* qp = QKV + (size_t)b * DIM_L * DIM_3C + 0 * DIM_C + g * HD;
  const bf16* kp = QKV + (size_t)b * DIM_L * DIM_3C + 1 * DIM_C + g * HD;
  const bf16* vp = QKV + (size_t)b * DIM_L * DIM_3C + 2 * DIM_C + g * HD;

  __shared__ bf16 qT[64 * TPAD];    // qT[d*TPAD + l] = q[l0+l, d]
  __shared__ bf16 kT[64 * TPAD];    // kT[e*TPAD + l] = k[l0+l, e]
  __shared__ float s_lds[64 * 65];  // S matrix, padded stride
  __shared__ bf16 p_lds[64 * 64];   // softmax(S) in bf16, row-major [d][e]

  const int wave = threadIdx.x >> 5;
  const int d0 = (wave >> 1) * 16;   // 0..48 : S row tile
  const int e0 = (wave & 1) * 32;    // 0/32  : S col tile (2 x 16)

  v8f s0 = (v8f)(0.0f), s1 = (v8f)(0.0f);

  // ---- S = q^T k over L, chunked through LDS (transpose on store) ----
  for (int l0 = 0; l0 < DIM_L; l0 += 128) {
#pragma unroll
    for (int i = 0; i < 4; ++i) {
      int idx = threadIdx.x + i * 256;   // 1024 chunks of 8 bf16 per matrix
      int r = idx >> 3;                  // l within chunk: 0..127
      int cc = (idx & 7) * 8;            // d/e start: 0,8,..,56
      v8bf qv = *(const v8bf*)&qp[(size_t)(l0 + r) * DIM_3C + cc];
      v8bf kv = *(const v8bf*)&kp[(size_t)(l0 + r) * DIM_3C + cc];
#pragma unroll
      for (int j = 0; j < 8; ++j) {
        qT[(cc + j) * TPAD + r] = qv[j];
        kT[(cc + j) * TPAD + r] = kv[j];
      }
    }
    __syncthreads();
#pragma unroll
    for (int kk = 0; kk < 128; kk += 32) {
      v16bf a  = load_fragA(&qT[d0 * TPAD + kk], TPAD);        // A[d,l]
      v16bf b0 = load_fragB(&kT[e0 * TPAD + kk], TPAD);        // B[l,e]
      v16bf b1 = load_fragB(&kT[(e0 + 16) * TPAD + kk], TPAD);
      s0 = wmma_bf16(a, b0, s0);
      s1 = wmma_bf16(a, b1, s1);
    }
    __syncthreads();
  }

  // ---- scale + dump S to LDS ----
  {
    const float scale = 0.015625f;  // 4096^-0.5
    int lane = threadIdx.x & 31;
    int n = lane & 15, hi = lane >> 4;
#pragma unroll
    for (int r = 0; r < 8; ++r) {
      s_lds[(d0 + hi * 8 + r) * 65 + e0 + n]      = s0[r] * scale;
      s_lds[(d0 + hi * 8 + r) * 65 + e0 + 16 + n] = s1[r] * scale;
    }
  }
  __syncthreads();

  // ---- softmax over e (one thread per row d), emit bf16 tile ----
  if (threadIdx.x < 64) {
    float* row = &s_lds[threadIdx.x * 65];
    float m = row[0];
#pragma unroll 4
    for (int j = 1; j < 64; ++j) m = fmaxf(m, row[j]);
    float sum = 0.0f;
#pragma unroll 4
    for (int j = 0; j < 64; ++j) { float e = __expf(row[j] - m); row[j] = e; sum += e; }
    float inv = 1.0f / sum;
    bf16* prow = &p_lds[threadIdx.x * 64];
#pragma unroll 4
    for (int j = 0; j < 64; ++j) prow[j] = (bf16)(row[j] * inv);
  }
  __syncthreads();

  // ---- out = v @ attn^T : M=L, N=d(64), K=e(64) ----
  v16bf bfr[2][4];
#pragma unroll
  for (int ks = 0; ks < 2; ++ks)
#pragma unroll
    for (int ni = 0; ni < 4; ++ni)
      bfr[ks][ni] = load_fragB(&p_lds[(ni * 16) * 64 + ks * 32], 64);

  const int lane = threadIdx.x & 31;
  const int n = lane & 15;
  const int hi = lane >> 4;
  for (int m0 = wave * 32; m0 < DIM_L; m0 += 256) {
    v8f o[2][4];
#pragma unroll
    for (int mi = 0; mi < 2; ++mi)
#pragma unroll
      for (int ni = 0; ni < 4; ++ni) o[mi][ni] = (v8f)(0.0f);
#pragma unroll
    for (int ks = 0; ks < 2; ++ks) {
      v16bf a0 = load_fragA(vp + (size_t)m0 * DIM_3C + ks * 32, DIM_3C);
      v16bf a1 = load_fragA(vp + (size_t)(m0 + 16) * DIM_3C + ks * 32, DIM_3C);
#pragma unroll
      for (int ni = 0; ni < 4; ++ni) {
        o[0][ni] = wmma_bf16(a0, bfr[ks][ni], o[0][ni]);
        o[1][ni] = wmma_bf16(a1, bfr[ks][ni], o[1][ni]);
      }
    }
#pragma unroll
    for (int mi = 0; mi < 2; ++mi)
#pragma unroll
      for (int ni = 0; ni < 4; ++ni)
#pragma unroll
        for (int r = 0; r < 8; ++r) {
          int l = m0 + mi * 16 + hi * 8 + r;
          int col = g * HD + ni * 16 + n;
          OutAtt[((size_t)b * DIM_L + l) * DIM_C + col] = (bf16)o[mi][ni][r];
        }
  }
}

// ---------------------------------------------------------------------------
// Launch.  Workspace layout (needs ~162 MB of d_ws):
//   Xbf    [32768 x  512] bf16   33,554,432 B
//   Wqkv_t [ 1536 x  512] bf16    1,572,864 B
//   Wproj_t[  512 x  512] bf16      524,288 B
//   QKV    [32768 x 1536] bf16  100,663,296 B
//   ATT    [32768 x  512] bf16   33,554,432 B
// ---------------------------------------------------------------------------
extern "C" void kernel_launch(void* const* d_in, const int* in_sizes, int n_in,
                              void* d_out, int out_size, void* d_ws, size_t ws_size,
                              hipStream_t stream) {
  (void)in_sizes; (void)n_in; (void)out_size; (void)ws_size;
  const float* x     = (const float*)d_in[0];
  const float* Wqkv  = (const float*)d_in[1];
  const float* bqkv  = (const float*)d_in[2];
  const float* Wproj = (const float*)d_in[3];
  const float* bproj = (const float*)d_in[4];
  const int*   szin  = (const int*)d_in[5];
  float* out = (float*)d_out;

  char* ws = (char*)d_ws;
  bf16* Xbf     = (bf16*)(ws);
  bf16* Wqkv_t  = (bf16*)(ws + 33554432);
  bf16* Wproj_t = (bf16*)(ws + 33554432 + 1572864);
  bf16* QKV     = (bf16*)(ws + 33554432 + 1572864 + 524288);
  bf16* ATT     = (bf16*)(ws + 33554432 + 1572864 + 524288 + 100663296);

  cvt_f32_bf16_v4<<<2048, 256, 0, stream>>>(x, Xbf, M_TOTAL * DIM_C / 4);
  transpose_cvt<<<1024, 256, 0, stream>>>(Wqkv, Wqkv_t, DIM_C, DIM_3C);
  transpose_cvt<<<512, 256, 0, stream>>>(Wproj, Wproj_t, DIM_C, DIM_C);

  // qkv = x @ W_qkv + b_qkv   (bf16 out)
  gemm_bf16_kernel<true><<<dim3(M_TOTAL / 64, DIM_3C / 256), 256, 0, stream>>>(
      Xbf, Wqkv_t, bqkv, QKV, nullptr, M_TOTAL, DIM_3C, DIM_C);

  // per-(b,g) cross-covariance attention
  attn_kernel<<<DIM_B * GROUPS, 256, 0, stream>>>(QKV, ATT);

  // out = att @ W_proj + b_proj  (f32 out)
  gemm_bf16_kernel<false><<<dim3(M_TOTAL / 64, DIM_C / 256), 256, 0, stream>>>(
      ATT, Wproj_t, bproj, nullptr, out, M_TOTAL, DIM_C, DIM_C);

  // tuple second element: pass `size` through bit-exactly
  copy_size_kernel<<<1, 1, 0, stream>>>(szin, ((int*)d_out) + M_TOTAL * DIM_C);
}
